// MultiheadAttention_56392920596705
// MI455X (gfx1250) — compile-verified, matrix operand order checked by reading, not run on hardware
//
#include <hip/hip_runtime.h>
#include <hip/hip_bf16.h>

// ---------------------------------------------------------------------------
// Causal multi-head attention forward for MI455X (gfx1250, wave32, WMMA, TDM).
//   B=4, S=2048, DIM=1024, HEADS=16, HEAD_DIM=64, SCALE=1/8.
// Pipeline:
//   1) qkv_proj : x[8192,1024] @ w_in^T[1024,3072] + b_in  -> Q/K/V f16 [B,H,S,64]
//   2) attn_fwd : flash-attention per (b,h,q-tile of 64), online softmax,
//                 v_wmma_f32_16x16x32_f16; K tiles staged by the Tensor Data
//                 Mover (tensor_load_to_lds + s_wait_tensorcnt)
//   3) out_proj : attn[8192,1024](f16) @ w_out^T + b_out -> f32 d_out
// ---------------------------------------------------------------------------

typedef _Float16 half16 __attribute__((ext_vector_type(16)));
typedef _Float16 half8  __attribute__((ext_vector_type(8)));
typedef float    f32x8  __attribute__((ext_vector_type(8)));
typedef float    f32x4  __attribute__((ext_vector_type(4)));
typedef unsigned int u32x4 __attribute__((ext_vector_type(4)));
typedef int      i32x4 __attribute__((ext_vector_type(4)));
typedef int      i32x8 __attribute__((ext_vector_type(8)));

#define B_    4
#define S_    2048
#define DIM_  1024
#define H_    16
#define HD_   64
#define M_    (B_ * S_)          // 8192 token rows
#define SCALE_ 0.125f
#define NEGF  (-1.0e30f)

// ---- WMMA wrapper: D = A(16x32 f16) * B(32x16 f16) + C(16x16 f32) ----------
__device__ __forceinline__ f32x8 wmma16(half16 a, half16 b, f32x8 c) {
  return __builtin_amdgcn_wmma_f32_16x16x32_f16(
      /*neg_a=*/false, a, /*neg_b=*/false, b,
      /*c_mod=*/(short)0, c, /*reuse_a=*/false, /*reuse_b=*/false);
}

// ---- Fragment load (A or B), 16 rows x 32 K, K-contiguous rows -------------
// ISA 7.12.2 16-bit A layout: lane<16 -> K {0..7,16..23}; lane>=16 -> {8..15,24..31};
// row (A: M, B: N) = lane & 15. Two 16-byte chunks per lane.
__device__ __forceinline__ half16 frag_ld(const _Float16* p0, int stride) {
  const int lane = threadIdx.x & 31;
  const _Float16* p = p0 + (size_t)(lane & 15) * stride + ((lane & 16) ? 8 : 0);
  half8 lo = *(const half8*)(p);
  half8 hi = *(const half8*)(p + 16);
  return __builtin_shufflevector(lo, hi, 0, 1, 2, 3, 4, 5, 6, 7,
                                 8, 9, 10, 11, 12, 13, 14, 15);
}

// ---- TDM: DMA a 64x64 f16 tile (row stride 64 halves in memory) into LDS ---
// Uses D# padding (interval 32 DWORDs = one 128B row, amount 4 DWORDs = 16B)
// so the landed LDS row stride is 72 halves, matching frag_ld's layout.
__device__ __forceinline__ void tdm_load_tile64(const _Float16* gsrc,
                                                _Float16* lds_dst) {
  const unsigned long long ga = (unsigned long long)(uintptr_t)gsrc;
  const unsigned lds = (unsigned)(uintptr_t)lds_dst;  // low 32 bits = LDS offset
  u32x4 g0;
  g0[0] = 1u;                                    // count=1 valid, user mode
  g0[1] = lds;                                   // lds_addr (bytes)
  g0[2] = (unsigned)ga;                          // global_addr[31:0]
  g0[3] = (unsigned)((ga >> 32) & 0x01FFFFFFu)   // global_addr[56:32]
          | (2u << 30);                          // type = 2 ("image")
  i32x8 g1;
  g1[0] = (1 << 16)          // data_size = 1 -> 2 bytes
        | (1 << 20)          // pad_enable
        | (4 << 22)          // pad_interval code 4 -> every 32 DWORDs
        | (3 << 25);         // pad_amount  code 3 -> 4 DWORDs
  g1[1] = (64 << 16);        // tensor_dim0 = 64 (bits 79:48, low half here)
  g1[2] = 0;                 // tensor_dim1[15:0] of (1<<20)
  g1[3] = 16 | (64 << 16);   // tensor_dim1[31:16]=16 (dim1=1M), tile_dim0=64
  g1[4] = 64;                // tile_dim1 = 64 rows, tile_dim2 = 0
  g1[5] = 64;                // tensor_dim0_stride = 64 elements
  g1[6] = 0;
  g1[7] = 0;
  i32x4 z = {0, 0, 0, 0};
#if __clang_major__ >= 23
  i32x8 z8 = {};
  __builtin_amdgcn_tensor_load_to_lds(g0, g1, z, z, z8, 0);
#else
  __builtin_amdgcn_tensor_load_to_lds(g0, g1, z, z, 0);
#endif
}

// ===========================================================================
// Kernel 1: QKV projection. C[m,n] = dot(x[m,:], w_in[n,:]) + b_in[n]
// Block tile 128(M) x 64(N), K-step 32, 8 waves (4x2), wave tile 32x32.
// Scatters f16 results into Q/K/V [B,H,S,64].
// ===========================================================================
__global__ __launch_bounds__(256) void qkv_proj_kernel(
    const float* __restrict__ x, const float* __restrict__ w_in,
    const float* __restrict__ b_in,
    _Float16* __restrict__ qh, _Float16* __restrict__ kh,
    _Float16* __restrict__ vh) {
  __shared__ __align__(16) _Float16 As[128 * 40];
  __shared__ __align__(16) _Float16 Bs[64 * 40];

  const int tid = threadIdx.x, lane = tid & 31, wave = tid >> 5;
  const int wm = wave >> 1, wn = wave & 1;
  const int m0 = blockIdx.y * 128, n0 = blockIdx.x * 64;

  f32x8 acc[2][2] = {};

  for (int k0 = 0; k0 < DIM_; k0 += 32) {
    { // stage A tile 128x32 (f32 -> f16)
      const int r = tid >> 1, c = (tid & 1) * 16;
      const float* src = x + (size_t)(m0 + r) * DIM_ + k0 + c;
      _Float16* dst = As + r * 40 + c;
#pragma unroll
      for (int j = 0; j < 16; j += 4) {
        f32x4 f = *(const f32x4*)(src + j);
        dst[j + 0] = (_Float16)f[0]; dst[j + 1] = (_Float16)f[1];
        dst[j + 2] = (_Float16)f[2]; dst[j + 3] = (_Float16)f[3];
      }
      if (k0 + 32 < DIM_) __builtin_prefetch(src + 32, 0, 3);
    }
    { // stage B tile 64x32 (rows of w_in = columns of w_in^T, K-contiguous)
      const int r = tid >> 2, c = (tid & 3) * 8;
      const float* src = w_in + (size_t)(n0 + r) * DIM_ + k0 + c;
      _Float16* dst = Bs + r * 40 + c;
#pragma unroll
      for (int j = 0; j < 8; j += 4) {
        f32x4 f = *(const f32x4*)(src + j);
        dst[j + 0] = (_Float16)f[0]; dst[j + 1] = (_Float16)f[1];
        dst[j + 2] = (_Float16)f[2]; dst[j + 3] = (_Float16)f[3];
      }
      if (k0 + 32 < DIM_) __builtin_prefetch(src + 32, 0, 3);
    }
    __syncthreads();

    half16 af[2], bf[2];
#pragma unroll
    for (int t = 0; t < 2; ++t) {
      af[t] = frag_ld(As + (wm * 32 + t * 16) * 40, 40);
      bf[t] = frag_ld(Bs + (wn * 32 + t * 16) * 40, 40);
    }
#pragma unroll
    for (int tm = 0; tm < 2; ++tm)
#pragma unroll
      for (int tn = 0; tn < 2; ++tn)
        acc[tm][tn] = wmma16(af[tm], bf[tn], acc[tm][tn]);
    __syncthreads();
  }

  // Epilogue: add bias, convert to f16, scatter to Q/K/V [B,H,S,64].
  const int hi8 = (lane & 16) ? 8 : 0;
#pragma unroll
  for (int tm = 0; tm < 2; ++tm)
#pragma unroll
    for (int tn = 0; tn < 2; ++tn)
#pragma unroll
      for (int r = 0; r < 8; ++r) {
        const int m = m0 + wm * 32 + tm * 16 + r + hi8;
        const int n = n0 + wn * 32 + tn * 16 + (lane & 15);
        const float val = acc[tm][tn][r] + b_in[n];
        const int bb = m >> 11, ss = m & (S_ - 1);
        const int part = n >> 10, c = n & (DIM_ - 1);
        const int hh = c >> 6, dd = c & 63;
        _Float16* dst = (part == 0) ? qh : ((part == 1) ? kh : vh);
        dst[(((size_t)bb * H_ + hh) * S_ + ss) * HD_ + dd] = (_Float16)val;
      }
}

// ===========================================================================
// Kernel 2: flash attention. Block = (q-tile of 64 rows, one (b,h)).
// 4 waves x 16 query rows. Key blocks of 64 stream through LDS; the K tile is
// staged by the Tensor Data Mover, V is transposed in by the waves.
// ===========================================================================
__global__ __launch_bounds__(128) void attn_fwd_kernel(
    const _Float16* __restrict__ qh, const _Float16* __restrict__ kh,
    const _Float16* __restrict__ vh, _Float16* __restrict__ ah) {
  __shared__ __align__(16) _Float16 Ks[64 * 72];       // K tile  [kpos][d]
  __shared__ __align__(16) _Float16 Vt[64 * 72];       // V tile^T [d][kpos]
  __shared__ __align__(16) _Float16 Ps[4 * 16 * 72];   // per-wave P tile [m][kpos]

  const int tid = threadIdx.x, lane = tid & 31, wave = tid >> 5;
  const int hi8 = (lane & 16) ? 8 : 0;
  const int bh = blockIdx.y, b = bh >> 4, h = bh & 15;
  const int qt = blockIdx.x, q0 = qt * 64, qw = q0 + wave * 16;
  const size_t base = (size_t)bh * S_ * HD_;

  // Q fragments (16 rows x 64 d) straight from global: rows d-contiguous.
  half16 qf[2];
  {
    const _Float16* qrow =
        qh + base + (size_t)(qw + (lane & 15)) * HD_ + ((lane & 16) ? 8 : 0);
#pragma unroll
    for (int kk = 0; kk < 2; ++kk) {
      half8 lo = *(const half8*)(qrow + kk * 32);
      half8 hi = *(const half8*)(qrow + kk * 32 + 16);
      qf[kk] = __builtin_shufflevector(lo, hi, 0, 1, 2, 3, 4, 5, 6, 7,
                                       8, 9, 10, 11, 12, 13, 14, 15);
    }
  }

  f32x8 o[4] = {};
  float m_i[8], l_i[8];
#pragma unroll
  for (int r = 0; r < 8; ++r) { m_i[r] = NEGF; l_i[r] = 0.f; }

  const int nkb = qt + 1;  // causal: key blocks 0..qt
  for (int kb = 0; kb < nkb; ++kb) {
    const int kb0 = kb * 64;
    __syncthreads();  // previous iteration's readers done with Ks/Vt

    // K tile via Tensor Data Mover (one descriptor, issued by wave 0 only).
    if (wave == 0)
      tdm_load_tile64(kh + base + (size_t)kb0 * HD_, Ks);

    { // cooperative stage: transposed V tile (TDM cannot transpose)
      const int r = tid >> 1, cb = (tid & 1) * 32;
      const _Float16* vsrc = vh + base + (size_t)(kb0 + r) * HD_ + cb;
#pragma unroll
      for (int j = 0; j < 32; j += 8) {
        half8 vv = *(const half8*)(vsrc + j);
#pragma unroll
        for (int e = 0; e < 8; ++e) Vt[(cb + j + e) * 72 + r] = (_Float16)vv[e];
      }
    }
    if (wave == 0) __builtin_amdgcn_s_wait_tensorcnt(0);
    __syncthreads();

    // S = Q.K^T : 16 x 64 scores = 4 N-tiles, 2 K-steps of 32 each.
    f32x8 s[4] = {};
#pragma unroll
    for (int n = 0; n < 4; ++n)
#pragma unroll
      for (int kk = 0; kk < 2; ++kk) {
        half16 bf = frag_ld(Ks + (n * 16) * 72 + kk * 32, 72);
        s[n] = wmma16(qf[kk], bf, s[n]);
      }

    // Online softmax (rows live across lanes 0..15 / 16..31 of one VGPR).
    const bool need_mask = (kb0 + 63 > qw);
    float p[4][8];
#pragma unroll
    for (int r = 0; r < 8; ++r) {
      const int qpos = qw + r + hi8;
      float mx = NEGF;
#pragma unroll
      for (int n = 0; n < 4; ++n) {
        float val = s[n][r] * SCALE_;
        if (need_mask) {
          const int kpos = kb0 + n * 16 + (lane & 15);
          if (kpos > qpos) val = NEGF;
        }
        p[n][r] = val;
        mx = fmaxf(mx, val);
      }
#pragma unroll
      for (int d = 1; d < 16; d <<= 1) mx = fmaxf(mx, __shfl_xor(mx, d, 16));
      const float m_new = fmaxf(m_i[r], mx);
      const float sf = __expf(m_i[r] - m_new);
      float rs = 0.f;
#pragma unroll
      for (int n = 0; n < 4; ++n) {
        p[n][r] = __expf(p[n][r] - m_new);
        rs += p[n][r];
      }
#pragma unroll
      for (int d = 1; d < 16; d <<= 1) rs += __shfl_xor(rs, d, 16);
      l_i[r] = l_i[r] * sf + rs;
      m_i[r] = m_new;
#pragma unroll
      for (int nd = 0; nd < 4; ++nd) o[nd][r] = o[nd][r] * sf;
    }

    // Re-pack P (C-layout) -> A-fragment layout via wave-private LDS tile.
    _Float16* pw = Ps + wave * (16 * 72);
#pragma unroll
    for (int r = 0; r < 8; ++r)
#pragma unroll
      for (int n = 0; n < 4; ++n)
        pw[(r + hi8) * 72 + n * 16 + (lane & 15)] = (_Float16)p[n][r];

    // O += P.V : 4 d-tiles, 2 K-steps of 32 key positions.
#pragma unroll
    for (int nd = 0; nd < 4; ++nd)
#pragma unroll
      for (int kk = 0; kk < 2; ++kk) {
        half16 a = frag_ld(pw + kk * 32, 72);
        half16 bv = frag_ld(Vt + (nd * 16) * 72 + kk * 32, 72);
        o[nd] = wmma16(a, bv, o[nd]);
      }
  }

  // Epilogue: normalize, write f16 attention output as [B,S,H*64] rows.
#pragma unroll
  for (int r = 0; r < 8; ++r) {
    const float inv = 1.f / l_i[r];
    const int qpos = qw + r + hi8;
    const size_t rowbase = ((size_t)b * S_ + qpos) * DIM_ + h * HD_;
#pragma unroll
    for (int nd = 0; nd < 4; ++nd)
      ah[rowbase + nd * 16 + (lane & 15)] = (_Float16)(o[nd][r] * inv);
  }
}

// ===========================================================================
// Kernel 3: output projection. out[m,n] = dot(attn[m,:], w_out[n,:]) + b_out[n]
// Same 128x64 tile GEMM; A already f16, W converted on the fly; f32 output.
// ===========================================================================
__global__ __launch_bounds__(256) void out_proj_kernel(
    const _Float16* __restrict__ ah, const float* __restrict__ w_out,
    const float* __restrict__ b_out, float* __restrict__ out) {
  __shared__ __align__(16) _Float16 As[128 * 40];
  __shared__ __align__(16) _Float16 Bs[64 * 40];

  const int tid = threadIdx.x, lane = tid & 31, wave = tid >> 5;
  const int wm = wave >> 1, wn = wave & 1;
  const int m0 = blockIdx.y * 128, n0 = blockIdx.x * 64;

  f32x8 acc[2][2] = {};

  for (int k0 = 0; k0 < DIM_; k0 += 32) {
    { // stage A tile (already f16): two 16-byte copies per thread
      const int r = tid >> 1, c = (tid & 1) * 16;
      const _Float16* src = ah + (size_t)(m0 + r) * DIM_ + k0 + c;
      _Float16* dst = As + r * 40 + c;
      *(half8*)(dst) = *(const half8*)(src);
      *(half8*)(dst + 8) = *(const half8*)(src + 8);
      if (k0 + 32 < DIM_) __builtin_prefetch(src + 32, 0, 3);
    }
    { // stage B tile (f32 -> f16)
      const int r = tid >> 2, c = (tid & 3) * 8;
      const float* src = w_out + (size_t)(n0 + r) * DIM_ + k0 + c;
      _Float16* dst = Bs + r * 40 + c;
#pragma unroll
      for (int j = 0; j < 8; j += 4) {
        f32x4 f = *(const f32x4*)(src + j);
        dst[j + 0] = (_Float16)f[0]; dst[j + 1] = (_Float16)f[1];
        dst[j + 2] = (_Float16)f[2]; dst[j + 3] = (_Float16)f[3];
      }
      if (k0 + 32 < DIM_) __builtin_prefetch(src + 32, 0, 3);
    }
    __syncthreads();

    half16 af[2], bf[2];
#pragma unroll
    for (int t = 0; t < 2; ++t) {
      af[t] = frag_ld(As + (wm * 32 + t * 16) * 40, 40);
      bf[t] = frag_ld(Bs + (wn * 32 + t * 16) * 40, 40);
    }
#pragma unroll
    for (int tm = 0; tm < 2; ++tm)
#pragma unroll
      for (int tn = 0; tn < 2; ++tn)
        acc[tm][tn] = wmma16(af[tm], bf[tn], acc[tm][tn]);
    __syncthreads();
  }

  const int hi8 = (lane & 16) ? 8 : 0;
#pragma unroll
  for (int tm = 0; tm < 2; ++tm)
#pragma unroll
    for (int tn = 0; tn < 2; ++tn)
#pragma unroll
      for (int r = 0; r < 8; ++r) {
        const int m = m0 + wm * 32 + tm * 16 + r + hi8;
        const int n = n0 + wn * 32 + tn * 16 + (lane & 15);
        out[(size_t)m * DIM_ + n] = acc[tm][tn][r] + b_out[n];
      }
}

// ===========================================================================
extern "C" void kernel_launch(void* const* d_in, const int* in_sizes, int n_in,
                              void* d_out, int out_size, void* d_ws,
                              size_t ws_size, hipStream_t stream) {
  const float* x     = (const float*)d_in[0];
  // d_in[1] = causal mask (int32) — causality is computed analytically.
  const float* w_in  = (const float*)d_in[2];
  const float* b_in  = (const float*)d_in[3];
  const float* w_out = (const float*)d_in[4];
  const float* b_out = (const float*)d_in[5];
  float* out = (float*)d_out;

  const size_t per = (size_t)B_ * H_ * S_ * HD_;  // 8,388,608 f16 elements
  _Float16* qh = (_Float16*)d_ws;
  _Float16* kh = qh + per;
  _Float16* vh = kh + per;
  _Float16* ah = vh + per;  // attention output, [B,S,DIM] f16

  qkv_proj_kernel<<<dim3(3 * DIM_ / 64, M_ / 128), 256, 0, stream>>>(
      x, w_in, b_in, qh, kh, vh);
  attn_fwd_kernel<<<dim3(S_ / 64, B_ * H_), 128, 0, stream>>>(qh, kh, vh, ah);
  out_proj_kernel<<<dim3(DIM_ / 64, M_ / 128), 256, 0, stream>>>(
      ah, w_out, b_out, out);
}
